// GambaBlock_21328807592169
// MI455X (gfx1250) — matrix operand order repaired and checked by reading, not compiled, not run on hardware
//
#include <hip/hip_runtime.h>

typedef __bf16 bf16_t;
typedef bf16_t v16bf __attribute__((ext_vector_type(16)));
typedef bf16_t v8bf  __attribute__((ext_vector_type(8)));
typedef float  v8f   __attribute__((ext_vector_type(8)));
typedef int    v4i   __attribute__((ext_vector_type(4)));

#define Bsz 4
#define Pch 512
#define C2c 256
#define Hh  32
#define Ww  32
#define Lq  1024
#define DIc 512
#define Nst 64
#define Rr  16
#define DHd 64
#define NH  4
#define Mrows (Bsz*Lq)   /* 4096 */
#define KCV 4
#define XDL 256          /* padded leading dim of x_dbl (real cols = 144) */

// ---- gfx1250 async global->LDS path (ASYNCcnt) — probe-confirmed -----------
#if defined(__has_builtin)
#  if __has_builtin(__builtin_amdgcn_global_load_async_to_lds_b128)
#    define ASYNC_LDS 1
#  endif
#  if __has_builtin(__builtin_amdgcn_s_wait_asynccnt)
#    define HAVE_WAIT_ASYNC 1
#  endif
#endif

static __device__ __forceinline__ void cp16(const bf16_t* g, bf16_t* l) {
#if defined(ASYNC_LDS)
  // confirmed: emits global_load_async_to_lds_b128 (ASYNCcnt-tracked)
  __builtin_amdgcn_global_load_async_to_lds_b128((v4i*)g, (v4i*)l, 0, 0);
#else
  *(v8bf*)l = *(const v8bf*)g;
#endif
}
static __device__ __forceinline__ void wait_stage() {
#if defined(ASYNC_LDS)
#  if defined(HAVE_WAIT_ASYNC)
  __builtin_amdgcn_s_wait_asynccnt(0);
#  else
  asm volatile("s_wait_asynccnt 0x0" ::: "memory");
#  endif
#endif
}

static __device__ __forceinline__ float sigf(float x){ return 1.f/(1.f+__expf(-x)); }

// ---------------------------------------------------------------------------
// Batched bf16 WMMA GEMM:  C[M,N] = act( beta*C + A[M,K] @ Bt^T[K,N] + bias )
// A row-major (M,K,lda).  Bt is N-major: Bt[n][k] (ldbt)  == B[k][n].
// C f32 row-major (ldc). M%128==0, N%128==0, K%32==0 guaranteed by caller.
// Block tile 128x128, 8 waves; wave = 32x64 strip (2x4 wmma accumulators,
// B-fragments reused across both A-fragments). Double-buffered LDS staged via
// async global->LDS DMA; branch-free steady-state loop (last iter peeled).
// ---------------------------------------------------------------------------
#define BM 128
#define BN 128
#define BK 32
#define LDP 40   /* padded LDS row stride (bank-conflict spread, 16B aligned) */

__global__ __launch_bounds__(256) void k_gemm(
    const bf16_t* __restrict__ A, const bf16_t* __restrict__ Bt,
    float* __restrict__ C,
    int K, int lda, int ldbt, int ldc,
    long long sA, long long sB, long long sC,
    const float* __restrict__ bias, int act, int beta)
{
  __shared__ bf16_t shA[2][BM][LDP];       // 2 x 10 KB
  __shared__ bf16_t shB[2][BN][LDP];       // 2 x 10 KB

  const int zb = blockIdx.z;
  A  += sA * zb; Bt += sB * zb; C += sC * zb;
  const long long m0 = (long long)blockIdx.y * BM;
  const long long n0 = (long long)blockIdx.x * BN;
  const int tid  = threadIdx.x;
  const int wave = tid >> 5;
  const int lane = tid & 31;
  const int wm = wave >> 1;            // 0..3 : 32-row strip
  const int wn = wave & 1;             // 0..1 : 64-col half
  const int lrow  = lane & 15;
  const int lhalf = lane >> 4;

  // staging: A and B tiles are both 128 rows x 32 cols; 2 x b128 per thread each
  const int sr = tid >> 1;             // 0..127
  const int sc = (tid & 1) << 4;       // 0,16

  const bf16_t* gA = A  + (m0 + sr) * lda  + sc;
  const bf16_t* gB = Bt + (n0 + sr) * ldbt + sc;

  v8f acc[2][4] = {};

  auto stage = [&](int k0, int buf) {
    cp16(gA + k0,     &shA[buf][sr][sc]);
    cp16(gA + k0 + 8, &shA[buf][sr][sc + 8]);
    cp16(gB + k0,     &shB[buf][sr][sc]);
    cp16(gB + k0 + 8, &shB[buf][sr][sc + 8]);
#if !defined(ASYNC_LDS)
    __builtin_prefetch((const void*)(gA + k0 + BK), 0, 1);
    __builtin_prefetch((const void*)(gB + k0 + BK), 0, 1);
#endif
  };

  auto compute = [&](int buf) {
    v16bf af[2];
    #pragma unroll
    for (int a = 0; a < 2; ++a) {
      const bf16_t* pa = &shA[buf][wm * 32 + a * 16 + lrow][lhalf * 8];
      v8bf lo = *(const v8bf*)pa;
      v8bf hi = *(const v8bf*)(pa + 16);
      #pragma unroll
      for (int i = 0; i < 8; ++i) { af[a][i] = lo[i]; af[a][8 + i] = hi[i]; }
    }
    #pragma unroll
    for (int t = 0; t < 4; ++t) {
      const bf16_t* pb = &shB[buf][wn * 64 + t * 16 + lrow][lhalf * 8];
      v8bf lo = *(const v8bf*)pb;
      v8bf hi = *(const v8bf*)(pb + 16);
      v16bf bfv;
      #pragma unroll
      for (int i = 0; i < 8; ++i) { bfv[i] = lo[i]; bfv[8 + i] = hi[i]; }
      acc[0][t] = __builtin_amdgcn_wmma_f32_16x16x32_bf16(
          false, af[0], false, bfv, (short)0, acc[0][t], false, false);
      acc[1][t] = __builtin_amdgcn_wmma_f32_16x16x32_bf16(
          false, af[1], false, bfv, (short)0, acc[1][t], false, false);
    }
  };

  const int nk = K / BK;
  stage(0, 0);
  wait_stage();
  __syncthreads();
  int cur = 0;
  for (int kt = 0; kt < nk - 1; ++kt) {   // branch-free steady state
    stage((kt + 1) * BK, cur ^ 1);
    compute(cur);
    wait_stage();          // our own next-tile stage ops are done
    __syncthreads();       // everyone finished reading cur + staging next
    cur ^= 1;
  }
  compute(cur);            // peeled last iteration

  #pragma unroll
  for (int a = 0; a < 2; ++a) {
    #pragma unroll
    for (int t = 0; t < 4; ++t) {
      long long col = n0 + wn * 64 + t * 16 + lrow;
      float bvv = bias ? bias[col] : 0.f;
      #pragma unroll
      for (int r = 0; r < 8; ++r) {
        long long row = m0 + wm * 32 + a * 16 + lhalf * 8 + r;
        float v = acc[a][t][r] + bvv;
        float* cp = C + row * ldc + col;
        if (beta) v += *cp;
        if (act == 1) v = v * sigf(v);        // silu
        else if (act == 2) v = sigf(v);       // sigmoid
        *cp = v;
      }
    }
  }
}

// ---------------------------------------------------------------------------
// Elementwise / helper kernels
// ---------------------------------------------------------------------------

// seq = x1(b,l,c) + pos(c,l);  x2seq(b,l,c);  zero guard rows around seq.
__global__ __launch_bounds__(256) void k_prep_seq(
    const float* __restrict__ x, const float* __restrict__ rhc,
    const float* __restrict__ rwc,
    bf16_t* __restrict__ seqbf, bf16_t* __restrict__ x2bf)
{
  long long idx = (long long)blockIdx.x * 256 + threadIdx.x;   // Mrows*C2c
  int c = (int)(idx % C2c);
  long long m = idx / C2c;
  int l = (int)(m % Lq);
  int b = (int)(m / Lq);
  float pos = rhc[c * Hh + (l & 31)] + rwc[c * Ww + (l >> 5)];
  float s = x[((long long)(b * Pch + c)) * Lq + l] + pos;
  seqbf[idx] = (bf16_t)s;
  x2bf[idx]  = (bf16_t)(x[((long long)(b * Pch + C2c + c)) * Lq + l]);
  if (idx < C2c) {
    seqbf[idx - (long long)C2c] = (bf16_t)0.f;
    seqbf[(long long)Mrows * C2c + idx] = (bf16_t)0.f;
  }
}

// convert weight to bf16 Bt layout (N-major, rows padded with zeros):
// out[n*Kk + k] = (n < Nreal) ? in[n*sN + k*sK] : 0
__global__ __launch_bounds__(256) void k_wcvt(
    const float* __restrict__ in, bf16_t* __restrict__ out,
    int Nreal, int Npad, int Kk, int sN, int sK)
{
  long long idx = (long long)blockIdx.x * 256 + threadIdx.x;
  if (idx >= (long long)Npad * Kk) return;
  int n = (int)(idx / Kk);
  int k = (int)(idx % Kk);
  out[idx] = (n < Nreal) ? (bf16_t)in[(long long)n * sN + (long long)k * sK]
                         : (bf16_t)0.f;
}

// depthwise causal conv (KC=4) over L on xi = xz[:, :DI], then SiLU.
__global__ __launch_bounds__(256) void k_conv_silu(
    const float* __restrict__ xz, const float* __restrict__ conv_w,
    const float* __restrict__ conv_b,
    float* __restrict__ xs, bf16_t* __restrict__ xsbf)
{
  long long idx = (long long)blockIdx.x * 256 + threadIdx.x;   // Mrows*DIc
  int di = (int)(idx % DIc);
  long long m = idx / DIc;
  int l = (int)(m % Lq);
  long long b = m / Lq;
  float s = conv_b[di];
  #pragma unroll
  for (int k = 0; k < KCV; ++k) {
    int ls = l - (KCV - 1) + k;
    if (ls >= 0) s += conv_w[di * KCV + k] * xz[(b * Lq + ls) * (2 * DIc) + di];
  }
  float v = s * sigf(s);
  xs[idx] = v;
  xsbf[idx] = (bf16_t)v;
}

// dt = softplus(xdbl[:, :R] @ W_dt^T + b_dt)   (xdbl leading dim = XDL)
__global__ __launch_bounds__(256) void k_dt(
    const float* __restrict__ xdbl, const float* __restrict__ Wdt,
    const float* __restrict__ bdt, float* __restrict__ dt)
{
  long long idx = (long long)blockIdx.x * 256 + threadIdx.x;   // Mrows*DIc
  int di = (int)(idx % DIc);
  long long m = idx / DIc;
  const float* xr = xdbl + m * XDL;
  float s = bdt[di];
  #pragma unroll
  for (int r = 0; r < Rr; ++r) s += xr[r] * Wdt[di * Rr + r];
  dt[idx] = (s > 20.f) ? s : __logf(1.f + __expf(s));
}

// selective scan: one wave32 per (b, di); lane holds states n and n+32.
__global__ __launch_bounds__(32) void k_scan(
    const float* __restrict__ dt, const float* __restrict__ xdbl,
    const float* __restrict__ xs, const float* __restrict__ A_log,
    float* __restrict__ ys)
{
  int bid = blockIdx.x;                  // Bsz*DIc
  int b  = bid >> 9;
  int di = bid & (DIc - 1);
  int lane = threadIdx.x;
  float A0 = -__expf(A_log[di * Nst + lane]);
  float A1 = -__expf(A_log[di * Nst + lane + 32]);
  float h0 = 0.f, h1 = 0.f;
  for (int l = 0; l < Lq; ++l) {
    long long m = (long long)b * Lq + l;
    float dv = dt[m * DIc + di];
    float u  = xs[m * DIc + di];
    const float* xr = xdbl + m * XDL;
    float B0 = xr[Rr + lane],        B1 = xr[Rr + lane + 32];
    float C0 = xr[Rr + Nst + lane],  C1 = xr[Rr + Nst + lane + 32];
    float du = dv * u;
    h0 = __expf(dv * A0) * h0 + du * B0;
    h1 = __expf(dv * A1) * h1 + du * B1;
    float y = h0 * C0 + h1 * C1;
    #pragma unroll
    for (int o = 16; o > 0; o >>= 1) y += __shfl_xor(y, o, 32);
    if (lane == 0) ys[m * DIc + di] = y;
  }
}

// ym_in = (ys + D*xs) * silu(z),  z = xz[:, DI:]
__global__ __launch_bounds__(256) void k_ycomb(
    const float* __restrict__ ys, const float* __restrict__ xs,
    const float* __restrict__ xz, const float* __restrict__ Dp,
    bf16_t* __restrict__ ymbf)
{
  long long idx = (long long)blockIdx.x * 256 + threadIdx.x;   // Mrows*DIc
  int di = (int)(idx % DIc);
  long long m = idx / DIc;
  float z = xz[m * (2 * DIc) + DIc + di];
  float v = (ys[idx] + Dp[di] * xs[idx]) * (z * sigf(z));
  ymbf[idx] = (bf16_t)v;
}

// a_in(b,lp,c) = (ym*ctx) through reference's raw reshape (b, c*L+lp)
__global__ __launch_bounds__(256) void k_ain(
    const float* __restrict__ ym, const float* __restrict__ ctx,
    bf16_t* __restrict__ ainbf)
{
  long long idx = (long long)blockIdx.x * 256 + threadIdx.x;   // Mrows*C2c
  int c = (int)(idx % C2c);
  long long m = idx / C2c;
  int lp = (int)(m % Lq);
  long long b = m / Lq;
  long long src = b * (long long)Lq * C2c + (long long)c * Lq + lp;
  ainbf[idx] = (bf16_t)(ym[src] * ctx[src]);
}

// pack q/k heads (e,i,d) and v transposed (e, d[128 pad], j); e = b*4+h
__global__ __launch_bounds__(256) void k_pack(
    const float* __restrict__ qg, const float* __restrict__ kg,
    const float* __restrict__ vg,
    bf16_t* __restrict__ qh, bf16_t* __restrict__ kh, bf16_t* __restrict__ vT)
{
  long long idx = (long long)blockIdx.x * 256 + threadIdx.x;   // 16*1024*64
  int e = (int)(idx >> 16);
  int rem = (int)(idx & 65535);
  int i = rem >> 6;
  int d = rem & 63;
  int b = e >> 2, h = e & 3;
  long long src = ((long long)(b * Lq + i)) * C2c + h * DHd + d;
  float qv = qg[src], kv = kg[src], vv = vg[src];
  qh[idx] = (bf16_t)qv;
  kh[idx] = (bf16_t)kv;
  long long t = ((long long)e * 128 + d) * Lq + i;
  vT[t] = (bf16_t)vv;
  vT[t + (long long)64 * Lq] = (bf16_t)0.f;   // zero pad rows d=64..127
}

// posmT(e,i,d) = rel_h_m[h,d,i%H] + rel_w_m[h,d,i/H], replicated over b
__global__ __launch_bounds__(256) void k_posm(
    const float* __restrict__ rhm, const float* __restrict__ rwm,
    bf16_t* __restrict__ posmT)
{
  long long idx = (long long)blockIdx.x * 256 + threadIdx.x;   // 16*1024*64
  int e = (int)(idx >> 16);
  int rem = (int)(idx & 65535);
  int i = rem >> 6;
  int d = rem & 63;
  int h = e & 3;
  float v = rhm[(h * DHd + d) * Hh + (i & 31)] + rwm[(h * DHd + d) * Ww + (i >> 5)];
  posmT[idx] = (bf16_t)v;
}

// row softmax (16384 rows of 1024) -> bf16 attn
__global__ __launch_bounds__(256) void k_softmax(
    const float* __restrict__ E, bf16_t* __restrict__ attn)
{
  long long row = blockIdx.x;
  const float* e = E + row * Lq;
  bf16_t* a = attn + row * Lq;
  __shared__ float red[8];
  int tid = threadIdx.x, lane = tid & 31, wave = tid >> 5;
  float mx = -3.4e38f;
  for (int j = tid; j < Lq; j += 256) mx = fmaxf(mx, e[j]);
  #pragma unroll
  for (int o = 16; o > 0; o >>= 1) mx = fmaxf(mx, __shfl_xor(mx, o, 32));
  if (lane == 0) red[wave] = mx;
  __syncthreads();
  float m2 = red[0];
  #pragma unroll
  for (int i = 1; i < 8; ++i) m2 = fmaxf(m2, red[i]);
  __syncthreads();
  float s = 0.f;
  for (int j = tid; j < Lq; j += 256) s += __expf(e[j] - m2);
  #pragma unroll
  for (int o = 16; o > 0; o >>= 1) s += __shfl_xor(s, o, 32);
  if (lane == 0) red[wave] = s;
  __syncthreads();
  float st = 0.f;
  #pragma unroll
  for (int i = 0; i < 8; ++i) st += red[i];
  float inv = 1.f / st;
  for (int j = tid; j < Lq; j += 256) a[j] = (bf16_t)(__expf(e[j] - m2) * inv);
}

// out_h(e,i,d[128]) -> b_in(b,lp, h*64+d)
__global__ __launch_bounds__(256) void k_unpack(
    const float* __restrict__ outh, bf16_t* __restrict__ binbf)
{
  long long idx = (long long)blockIdx.x * 256 + threadIdx.x;   // 16*1024*64
  int e = (int)(idx >> 16);
  int rem = (int)(idx & 65535);
  int i = rem >> 6;
  int d = rem & 63;
  int b = e >> 2, h = e & 3;
  binbf[((long long)(b * Lq + i)) * C2c + h * DHd + d] =
      (bf16_t)outh[((long long)e * Lq + i) * 128 + d];
}

// final gating: one wave per row m; out (B,P,H,W)
__global__ __launch_bounds__(256) void k_final(
    const float* __restrict__ aP, const float* __restrict__ bP,
    const float* __restrict__ Wg, const float* __restrict__ bg,
    float* __restrict__ out)
{
  int wave = threadIdx.x >> 5, lane = threadIdx.x & 31;
  long long m = (long long)blockIdx.x * 8 + wave;
  const float* ar = aP + m * Pch;
  const float* br = bP + m * Pch;
  float sa = 0.f, sb = 0.f;
  #pragma unroll
  for (int i = 0; i < 16; ++i) {
    int p = lane + i * 32;
    float w = Wg[p];
    sa += w * ar[p];
    sb += w * br[p];
  }
  #pragma unroll
  for (int o = 16; o > 0; o >>= 1) { sa += __shfl_xor(sa, o, 32); sb += __shfl_xor(sb, o, 32); }
  float ga = sigf(sa + bg[0]);
  float gb = sigf(sb + bg[0]);
  int b = (int)(m / Lq), l = (int)(m % Lq);
  #pragma unroll
  for (int i = 0; i < 16; ++i) {
    int p = lane + i * 32;
    out[((long long)(b * Pch + p)) * Lq + l] = ga * ar[p] + gb * br[p];
  }
}

// ---------------------------------------------------------------------------
// Host
// ---------------------------------------------------------------------------
static void gemm_launch(hipStream_t s, const bf16_t* A, const bf16_t* Bt, float* C,
    int M, int N, int K, int lda, int ldbt, int ldc,
    long long sA, long long sB, long long sC, int batches,
    const float* bias, int act, int beta)
{
  dim3 g(N / BN, M / BM, batches);
  k_gemm<<<g, 256, 0, s>>>(A, Bt, C, K, lda, ldbt, ldc, sA, sB, sC, bias, act, beta);
}

extern "C" void kernel_launch(void* const* d_in, const int* in_sizes, int n_in,
                              void* d_out, int out_size, void* d_ws, size_t ws_size,
                              hipStream_t stream)
{
  (void)in_sizes; (void)n_in; (void)out_size; (void)ws_size;
  const float* x      = (const float*)d_in[0];
  const float* rhc    = (const float*)d_in[1];
  const float* rwc    = (const float*)d_in[2];
  const float* gate_w = (const float*)d_in[3];
  const float* gate_b = (const float*)d_in[4];
  const float* W_in   = (const float*)d_in[5];
  const float* conv_w = (const float*)d_in[6];
  const float* conv_b = (const float*)d_in[7];
  const float* W_x    = (const float*)d_in[8];
  const float* W_dt   = (const float*)d_in[9];
  const float* b_dt   = (const float*)d_in[10];
  const float* A_log  = (const float*)d_in[11];
  const float* D_p    = (const float*)d_in[12];
  const float* W_out  = (const float*)d_in[13];
  const float* Wq     = (const float*)d_in[14];
  const float* bq     = (const float*)d_in[15];
  const float* Wk     = (const float*)d_in[16];
  const float* bk     = (const float*)d_in[17];
  const float* Wv     = (const float*)d_in[18];
  const float* bv     = (const float*)d_in[19];
  const float* rhm    = (const float*)d_in[20];
  const float* rwm    = (const float*)d_in[21];
  // d_in[22]=reg_qk, d_in[23]=reg_v: reg head is sliced away by out[:, :HEAD];
  // per-head softmax => it cannot affect the output. Skipped.
  const float* Wa     = (const float*)d_in[24];
  const float* ba     = (const float*)d_in[25];
  const float* Wb     = (const float*)d_in[26];
  const float* bb     = (const float*)d_in[27];
  const float* Wg     = (const float*)d_in[28];
  const float* bg     = (const float*)d_in[29];
  float* out = (float*)d_out;

  char* base = (char*)d_ws;
  size_t off = 0;
  auto alloc = [&](size_t elems, size_t esz) -> void* {
    void* p = base + off;
    off += (elems * esz + 255) & ~(size_t)255;
    return p;
  };

  bf16_t* seq_raw = (bf16_t*)alloc((size_t)Mrows * C2c + 2 * C2c, 2);
  bf16_t* seqbf   = seq_raw + C2c;                       // guard row each side
  bf16_t* x2bf    = (bf16_t*)alloc((size_t)Mrows * C2c, 2);
  // weights, Bt (= original W) layout, bf16
  bf16_t* W_in_bf = (bf16_t*)alloc((size_t)2 * DIc * C2c, 2);
  bf16_t* gate_bf = (bf16_t*)alloc((size_t)3 * C2c * C2c, 2);
  bf16_t* Wq_bf   = (bf16_t*)alloc((size_t)C2c * C2c, 2);
  bf16_t* Wk_bf   = (bf16_t*)alloc((size_t)C2c * C2c, 2);
  bf16_t* Wv_bf   = (bf16_t*)alloc((size_t)C2c * C2c, 2);
  bf16_t* Wx_bf   = (bf16_t*)alloc((size_t)XDL * DIc, 2);       // padded 144->256 rows
  bf16_t* Wout_bf = (bf16_t*)alloc((size_t)C2c * DIc, 2);
  bf16_t* Wa_bf   = (bf16_t*)alloc((size_t)Pch * C2c, 2);
  bf16_t* Wb_bf   = (bf16_t*)alloc((size_t)Pch * C2c, 2);
  float*  xz      = (float*) alloc((size_t)Mrows * 2 * DIc, 4);
  float*  ctx     = (float*) alloc((size_t)Mrows * C2c, 4);
  float*  xs      = (float*) alloc((size_t)Mrows * DIc, 4);
  bf16_t* xsbf    = (bf16_t*)alloc((size_t)Mrows * DIc, 2);
  float*  xdbl    = (float*) alloc((size_t)Mrows * XDL, 4);
  float*  dtb     = (float*) alloc((size_t)Mrows * DIc, 4);
  float*  ys      = (float*) alloc((size_t)Mrows * DIc, 4);
  bf16_t* ymbf    = (bf16_t*)alloc((size_t)Mrows * DIc, 2);
  float*  ym      = (float*) alloc((size_t)Mrows * C2c, 4);
  bf16_t* ainbf   = (bf16_t*)alloc((size_t)Mrows * C2c, 2);
  float*  qg      = (float*) alloc((size_t)Mrows * C2c, 4);
  float*  kg      = (float*) alloc((size_t)Mrows * C2c, 4);
  float*  vg      = (float*) alloc((size_t)Mrows * C2c, 4);
  size_t  nhd     = (size_t)16 * Lq * DHd;               // 1,048,576
  bf16_t* qh      = (bf16_t*)alloc(nhd, 2);
  bf16_t* kh      = (bf16_t*)alloc(nhd, 2);
  bf16_t* vT      = (bf16_t*)alloc((size_t)16 * 128 * Lq, 2);   // d padded to 128
  bf16_t* posmT   = (bf16_t*)alloc(nhd, 2);
  float*  energy  = (float*) alloc((size_t)16 * Lq * Lq, 4);
  bf16_t* attnbf  = (bf16_t*)alloc((size_t)16 * Lq * Lq, 2);
  float*  outh    = (float*) alloc((size_t)16 * Lq * 128, 4);   // ldc=128 padded
  bf16_t* binbf   = (bf16_t*)alloc((size_t)Mrows * C2c, 2);
  float*  aP      = (float*) alloc((size_t)Mrows * Pch, 4);
  float*  bP      = (float*) alloc((size_t)Mrows * Pch, 4);

  const int thr = 256;
  long long nMC = (long long)Mrows * C2c;                // 1,048,576
  long long nMD = (long long)Mrows * DIc;                // 2,097,152

  // ---- prep ----
  k_prep_seq<<<(unsigned)(nMC / thr), thr, 0, stream>>>(x, rhc, rwc, seqbf, x2bf);

  auto wcvt = [&](const float* in, bf16_t* o, int Nreal, int Npad, int Kk, int sN, int sK) {
    long long n = (long long)Npad * Kk;
    k_wcvt<<<(unsigned)((n + thr - 1) / thr), thr, 0, stream>>>(in, o, Nreal, Npad, Kk, sN, sK);
  };
  wcvt(W_in,  W_in_bf, 2 * DIc, 2 * DIc, C2c, C2c, 1);
  for (int t = 0; t < 3; ++t)
    wcvt(gate_w + t, gate_bf + (size_t)t * C2c * C2c, C2c, C2c, C2c, C2c * 3, 3);
  wcvt(Wq,    Wq_bf,   C2c, C2c, C2c, C2c, 1);
  wcvt(Wk,    Wk_bf,   C2c, C2c, C2c, C2c, 1);
  wcvt(Wv,    Wv_bf,   C2c, C2c, C2c, C2c, 1);
  wcvt(W_x,   Wx_bf,   144, XDL, DIc, DIc, 1);           // pad rows 144..255 = 0
  wcvt(W_out, Wout_bf, C2c, C2c, DIc, DIc, 1);
  wcvt(Wa,    Wa_bf,   Pch, Pch, C2c, C2c, 1);
  wcvt(Wb,    Wb_bf,   Pch, Pch, C2c, C2c, 1);

  // ---- mamba branch ----
  gemm_launch(stream, seqbf, W_in_bf, xz, Mrows, 2 * DIc, C2c, C2c, C2c, 2 * DIc,
              0, 0, 0, 1, nullptr, 0, 0);
  // gate conv (3 taps via shifted A), sigmoid on last pass
  gemm_launch(stream, seqbf - C2c, gate_bf,                 ctx, Mrows, C2c, C2c, C2c, C2c, C2c,
              0, 0, 0, 1, gate_b, 0, 0);
  gemm_launch(stream, seqbf,       gate_bf + 1 * C2c * C2c, ctx, Mrows, C2c, C2c, C2c, C2c, C2c,
              0, 0, 0, 1, nullptr, 0, 1);
  gemm_launch(stream, seqbf + C2c, gate_bf + 2 * C2c * C2c, ctx, Mrows, C2c, C2c, C2c, C2c, C2c,
              0, 0, 0, 1, nullptr, 2, 1);
  k_conv_silu<<<(unsigned)(nMD / thr), thr, 0, stream>>>(xz, conv_w, conv_b, xs, xsbf);
  gemm_launch(stream, xsbf, Wx_bf, xdbl, Mrows, XDL, DIc, DIc, DIc, XDL,
              0, 0, 0, 1, nullptr, 0, 0);
  k_dt<<<(unsigned)(nMD / thr), thr, 0, stream>>>(xdbl, W_dt, b_dt, dtb);
  k_scan<<<Bsz * DIc, 32, 0, stream>>>(dtb, xdbl, xs, A_log, ys);
  k_ycomb<<<(unsigned)(nMD / thr), thr, 0, stream>>>(ys, xs, xz, D_p, ymbf);
  gemm_launch(stream, ymbf, Wout_bf, ym, Mrows, C2c, DIc, DIc, DIc, C2c,
              0, 0, 0, 1, nullptr, 0, 0);
  k_ain<<<(unsigned)(nMC / thr), thr, 0, stream>>>(ym, ctx, ainbf);

  // ---- attention branch ----
  gemm_launch(stream, x2bf, Wq_bf, qg, Mrows, C2c, C2c, C2c, C2c, C2c, 0, 0, 0, 1, bq, 0, 0);
  gemm_launch(stream, x2bf, Wk_bf, kg, Mrows, C2c, C2c, C2c, C2c, C2c, 0, 0, 0, 1, bk, 0, 0);
  gemm_launch(stream, x2bf, Wv_bf, vg, Mrows, C2c, C2c, C2c, C2c, C2c, 0, 0, 0, 1, bv, 0, 0);
  long long npk = (long long)nhd;
  k_pack<<<(unsigned)(npk / thr), thr, 0, stream>>>(qg, kg, vg, qh, kh, vT);
  k_posm<<<(unsigned)(npk / thr), thr, 0, stream>>>(rhm, rwm, posmT);
  // energy = q^T k + posm^T q  (batched over e = b*4+h); Bt operands are (e,i,d)
  gemm_launch(stream, qh,    kh, energy, Lq, Lq, DHd, DHd, DHd, Lq,
              65536, 65536, (long long)Lq * Lq, 16, nullptr, 0, 0);
  gemm_launch(stream, posmT, qh, energy, Lq, Lq, DHd, DHd, DHd, Lq,
              65536, 65536, (long long)Lq * Lq, 16, nullptr, 0, 1);
  k_softmax<<<16 * Lq, 256, 0, stream>>>(energy, attnbf);
  // out_h = attn @ v   (Bt = vT, N padded to 128)
  gemm_launch(stream, attnbf, vT, outh, Lq, 128, Lq, Lq, Lq, 128,
              (long long)Lq * Lq, (long long)128 * Lq, (long long)128 * Lq, 16,
              nullptr, 0, 0);
  k_unpack<<<(unsigned)(npk / thr), thr, 0, stream>>>(outh, binbf);

  // ---- fusion ----
  gemm_launch(stream, ainbf, Wa_bf, aP, Mrows, Pch, C2c, C2c, C2c, Pch,
              0, 0, 0, 1, ba, 0, 0);
  gemm_launch(stream, binbf, Wb_bf, bP, Mrows, Pch, C2c, C2c, C2c, Pch,
              0, 0, 0, 1, bb, 0, 0);
  k_final<<<Mrows / 8, 256, 0, stream>>>(aP, bP, Wg, bg, out);
}
